// LSTM_91216515432751
// MI455X (gfx1250) — 8-bit matrix operand order repaired and checked by reading, compiled, not run on hardware
//
#include <hip/hip_runtime.h>

typedef __attribute__((ext_vector_type(16))) int   v16i;
typedef __attribute__((ext_vector_type(8)))  float v8f;

namespace {
constexpr int kB = 512, kT = 784, kH = 256, kC = 10;
constexpr int kRows     = 16;
constexpr int kWaves    = 16;
constexpr int kThreads  = kWaves * 32;
constexpr int kH8Stride = 272;
constexpr int kPackedBytes = 4 * kH * kH;
}

__device__ __forceinline__ unsigned char f32_to_e4m3_sw(float x) {
  unsigned char s = 0;
  if (x < 0.0f) { s = 0x80; x = -x; }
  if (!(x == x)) return 0x7F;
  if (x >= 448.0f) return (unsigned char)(s | 0x7E);
  if (x < 0.0009765625f) return s;
  int e; float m = frexpf(x, &e);
  if (e >= -5) {
    int E = e + 6;
    int q = (int)rintf(m * 16.0f);
    if (q == 16) { q = 8; ++E; }
    if (E > 15 || (E == 15 && q > 14)) return (unsigned char)(s | 0x7E);
    return (unsigned char)(s | (E << 3) | (q - 8));
  }
  int d = (int)rintf(x * 512.0f);
  if (d > 7) return (unsigned char)(s | 0x08);
  return (unsigned char)(s | d);
}

__device__ __forceinline__ unsigned char f32_to_e4m3(float x) {
#if __has_builtin(__builtin_amdgcn_cvt_pk_fp8_f32)
  return (unsigned char)__builtin_amdgcn_cvt_pk_fp8_f32(x, x, 0, false);
#else
  return f32_to_e4m3_sw(x);
#endif
}

__device__ __forceinline__ float fast_rcp(float v) {
#if __has_builtin(__builtin_amdgcn_rcpf)
  return __builtin_amdgcn_rcpf(v);
#else
  return 1.0f / v;
#endif
}
__device__ __forceinline__ float sigmoidf_fast(float v) {
  return fast_rcp(1.0f + __expf(-v));
}
__device__ __forceinline__ float tanhf_fast(float v) {
#if __has_builtin(__builtin_amdgcn_tanhf)
  return __builtin_amdgcn_tanhf(v);
#else
  return 2.0f * sigmoidf_fast(2.0f * v) - 1.0f;
#endif
}

__global__ void pack_wh_fp8(const float* __restrict__ Wh,
                            unsigned char* __restrict__ out) {
  unsigned u  = blockIdx.x * 256u + threadIdx.x;
  unsigned b  = u & 15u;
  unsigned l  = (u >> 4) & 31u;
  unsigned p  = (u >> 9) & 3u;
  unsigned nt = (u >> 11) & 15u;
  unsigned kc = (u >> 15) & 1u;
  unsigned k  = u >> 16;
  unsigned K  = kc * 128u + p * 32u + ((l < 16u) ? 0u : 16u) + b;
  unsigned N  = nt * 16u + (l & 15u);
  unsigned tile = (k * 2u + kc) * 16u + nt;
  out[tile * 2048u + (p * 32u + l) * 16u + b] =
      f32_to_e4m3(Wh[k * 65536u + K * 256u + N]);
}

__global__ __launch_bounds__(kThreads, 1)
void lstm_persistent(const float* __restrict__ x,
                     const float* __restrict__ Wx,
                     const float* __restrict__ Wp,
                     const float* __restrict__ biasG,
                     const float* __restrict__ biasP,
                     const unsigned char* __restrict__ wpk,
                     float* __restrict__ out) {
  __shared__ __align__(16) unsigned char h8[2][kRows * kH8Stride];
  __shared__ float xbuf[2][kRows];
  __shared__ float hfin[kRows * kH];

  const int tid  = threadIdx.x;
  const int wave = tid >> 5;
  const int lane = tid & 31;
  const int lrow = lane & 15;
  const int hh   = lane >> 4;
  const int b0   = blockIdx.x * kRows;
  const int ncol = wave * 16 + lrow;

  union BF { v16i v; float4 q[4]; };
  BF bf[4][2];
  #pragma unroll
  for (int k = 0; k < 4; ++k)
    #pragma unroll
    for (int kc = 0; kc < 2; ++kc) {
      const unsigned char* tbase = wpk + ((k * 2 + kc) * 16 + wave) * 2048;
      #pragma unroll
      for (int p = 0; p < 4; ++p)
        bf[k][kc].q[p] = *(const float4*)(tbase + (p * 32 + lane) * 16);
    }

  v8f biasv[4];
  float wxv[4];
  #pragma unroll
  for (int k = 0; k < 4; ++k) {
    wxv[k] = Wx[k * kH + ncol];
    #pragma unroll
    for (int r = 0; r < 8; ++r) {
      const int m = r + hh * 8;
      biasv[k][r] = biasG[(k * kB + (b0 + m)) * kH + ncol];
    }
  }

  v8f cst = {};

  for (int i = tid; i < kRows * kH8Stride; i += kThreads) h8[0][i] = 0;
  if (tid < kRows) xbuf[0][tid] = x[(b0 + tid) * kT + 0];
  __syncthreads();

  for (int t = 0; t < kT; ++t) {
    const int cur = t & 1, nxt = cur ^ 1;

    union AF { v16i v; unsigned long long d[8]; };
    AF a0, a1;
    const unsigned char* hrow = &h8[cur][lrow * kH8Stride + hh * 16];
    #pragma unroll
    for (int c = 0; c < 8; ++c) {
      a0.d[c] = *(const unsigned long long*)(hrow + 32 * (c >> 1) + 8 * (c & 1));
      a1.d[c] = *(const unsigned long long*)(hrow + 32 * (c >> 1) + 8 * (c & 1) + 128);
    }

    if (wave == 0) {
      const int tnext = (t + 1 < kT) ? (t + 1) : (kT - 1);
      unsigned ldsoff = (unsigned)(size_t)&xbuf[nxt][lrow];
      unsigned goff   = ((unsigned)(b0 + lrow) * (unsigned)kT + (unsigned)tnext) * 4u;
      asm volatile("global_load_async_to_lds_b32 %0, %1, %2"
                   :: "v"(ldsoff), "v"(goff), "s"(x) : "memory");
    }

    v8f acc[4];
    #pragma unroll
    for (int k = 0; k < 4; ++k) {
      acc[k] = __builtin_amdgcn_wmma_f32_16x16x128_fp8_fp8(
          a0.v, bf[k][0].v, (short)0, biasv[k], false, false);
      acc[k] = __builtin_amdgcn_wmma_f32_16x16x128_fp8_fp8(
          a1.v, bf[k][1].v, (short)0, acc[k], false, false);
    }

    const bool last = (t == kT - 1);
    #pragma unroll
    for (int r = 0; r < 8; ++r) {
      const int m  = r + hh * 8;
      const float xm = xbuf[cur][m];
      const float g0 = sigmoidf_fast(acc[0][r] + xm * wxv[0]);
      const float g1 = sigmoidf_fast(acc[1][r] + xm * wxv[1]);
      const float g2 = sigmoidf_fast(acc[2][r] + xm * wxv[2]);
      const float g3 = sigmoidf_fast(acc[3][r] + xm * wxv[3]);
      const float cv = g0 * g1 + cst[r] * g2;
      cst[r] = cv;
      const float hv = tanhf_fast(cv) * g3;
      h8[nxt][m * kH8Stride + ncol] = f32_to_e4m3(hv);
      if (last) hfin[m * kH + ncol] = hv;
    }

    if (wave == 0) {
      asm volatile("s_wait_asynccnt 0x0" ::: "memory");
    }
    __syncthreads();
  }

  if (tid < kRows * kC) {
    const int r = tid / kC, cc = tid % kC;
    float s = biasP[(b0 + r) * kC + cc];
    for (int i = 0; i < kH; ++i) s += hfin[r * kH + i] * Wp[i * kC + cc];
    out[(b0 + r) * kC + cc] = s;
  }
}

extern "C" void kernel_launch(void* const* d_in, const int* in_sizes, int n_in,
                              void* d_out, int out_size, void* d_ws, size_t ws_size,
                              hipStream_t stream) {
  (void)in_sizes; (void)n_in; (void)out_size; (void)ws_size;
  const float* x   = (const float*)d_in[0];
  const float* Wx  = (const float*)d_in[1];
  const float* Wh  = (const float*)d_in[2];
  const float* Wp  = (const float*)d_in[3];
  const float* bG  = (const float*)d_in[4];
  const float* bP  = (const float*)d_in[5];
  unsigned char* wpk = (unsigned char*)d_ws;

  pack_wh_fp8<<<kPackedBytes / 256, 256, 0, stream>>>(Wh, wpk);
  lstm_persistent<<<kB / kRows, kThreads, 0, stream>>>(
      x, Wx, Wp, bG, bP, wpk, (float*)d_out);
}
